// ScaledMNISTNet_14568529068118
// MI455X (gfx1250) — matrix operand order repaired and checked by reading, not compiled, not run on hardware
//
#include <hip/hip_runtime.h>
#include <hip/hip_fp16.h>
#include <math.h>

#define DIMS 48
#define PDIM 50
#define NB   4
#define VOL  (DIMS*DIMS*DIMS)      /* 110592 */
#define PVOL (PDIM*PDIM*PDIM)      /* 125000 */
#define TOT  (NB*VOL)              /* 442368 */
#define NTAP 27
#define KTOT 864                   /* 32 ch * 27 taps */
#define EPSV 1e-5f

typedef __attribute__((ext_vector_type(16))) _Float16 v16h;
typedef __attribute__((ext_vector_type(8)))  float    v8f;

// ---------------- utility kernels ----------------

__global__ void zero_kernel(float* __restrict__ p, int n) {
    int i = blockIdx.x * blockDim.x + threadIdx.x;
    if (i < n) p[i] = 0.f;
}

// pad 48^3 -> 50^3 with zero border, per batch (C=1)
__global__ void pad_kernel(const float* __restrict__ in, float* __restrict__ out) {
    int i = blockIdx.x * blockDim.x + threadIdx.x;
    if (i >= NB * PVOL) return;
    int x = i % PDIM, y = (i / PDIM) % PDIM, z = (i / (PDIM*PDIM)) % PDIM, b = i / PVOL;
    float v = 0.f;
    if (x > 0 && x < PDIM-1 && y > 0 && y < PDIM-1 && z > 0 && z < PDIM-1)
        v = in[(size_t)b * VOL + (z-1)*(DIMS*DIMS) + (y-1)*DIMS + (x-1)];
    out[i] = v;
}

// ---------------- deformable conv stage (fused) ----------------
// Per output point: 81ch offset conv + 27ch mask conv (3^3 taps each, from the
// padded buffer), 27 deformed trilinear samples, modulation, 27-tap combine.
// Also accumulates sum / sumsq for the whole-tensor BN, and optionally writes
// xxx (=y) and p (clipped positions) for the second iteration.
__global__ __launch_bounds__(256) void deform_kernel(
        const float* __restrict__ pad,
        const float* __restrict__ w_p, const float* __restrict__ b_p,
        const float* __restrict__ w_m, const float* __restrict__ b_m,
        const float* __restrict__ w_dc,
        float* __restrict__ y_out, float* __restrict__ stat2,
        float* __restrict__ xxx_out, float* __restrict__ p_out) {
    __shared__ float swp[81*27];
    __shared__ float swm[27*27];
    __shared__ float swdc[27];
    __shared__ float sbp[81];
    __shared__ float sbm[27];
    __shared__ float red[256];
    int tid = threadIdx.x;
    for (int i = tid; i < 81*27; i += 256) swp[i] = w_p[i];
    for (int i = tid; i < 27*27; i += 256) swm[i] = w_m[i];
    if (tid < 27) swdc[tid] = w_dc[tid];
    if (tid < 81) sbp[tid]  = b_p[tid];
    if (tid < 27) sbm[tid]  = b_m[tid];
    __syncthreads();

    int idx = blockIdx.x * 256 + tid;     // grid is exactly TOT/256 blocks
    int xq = idx % DIMS, yq = (idx / DIMS) % DIMS,
        zq = (idx / (DIMS*DIMS)) % DIMS, bq = idx / VOL;
    const float* xp = pad + (size_t)bq * PVOL;

    float v[27];
    #pragma unroll
    for (int dz = 0; dz < 3; ++dz)
        #pragma unroll
        for (int dy = 0; dy < 3; ++dy)
            #pragma unroll
            for (int dx = 0; dx < 3; ++dx)
                v[dz*9 + dy*3 + dx] = xp[(zq+dz)*(PDIM*PDIM) + (yq+dy)*PDIM + (xq+dx)];

    const float PM1 = (float)(PDIM - 1);
    float acc = 0.f;
    for (int n = 0; n < NTAP; ++n) {
        float od = sbp[n], oh = sbp[27+n], ow = sbp[54+n], tm = sbm[n];
        #pragma unroll
        for (int t = 0; t < 27; ++t) {
            od = fmaf(swp[n*27 + t],      v[t], od);
            oh = fmaf(swp[(27+n)*27 + t], v[t], oh);
            ow = fmaf(swp[(54+n)*27 + t], v[t], ow);
            tm = fmaf(swm[n*27 + t],      v[t], tm);
        }
        float mn = 1.f / (1.f + expf(-tm));
        // p0 + pn = (coord+1) + (tapidx-1) = coord + tapidx
        float pdv = (float)(zq + (n/9))      + od;
        float phv = (float)(yq + ((n/3)%3))  + oh;
        float pwv = (float)(xq + (n%3))      + ow;
        float d0 = fminf(fmaxf(floorf(pdv), 0.f), PM1); float d1 = fminf(d0 + 1.f, PM1);
        float h0 = fminf(fmaxf(floorf(phv), 0.f), PM1); float h1 = fminf(h0 + 1.f, PM1);
        float w0 = fminf(fmaxf(floorf(pwv), 0.f), PM1); float w1 = fminf(w0 + 1.f, PM1);
        float pdc = fminf(fmaxf(pdv, 0.f), PM1);
        float phc = fminf(fmaxf(phv, 0.f), PM1);
        float pwc = fminf(fmaxf(pwv, 0.f), PM1);
        float wd0 = 1.f + d0 - pdc, wd1 = 1.f - d1 + pdc;
        float wh0 = 1.f + h0 - phc, wh1 = 1.f - h1 + phc;
        float ww0 = 1.f + w0 - pwc, ww1 = 1.f - w1 + pwc;
        int id0 = (int)d0, id1 = (int)d1, ih0 = (int)h0, ih1 = (int)h1,
            iw0 = (int)w0, iw1 = (int)w1;
        float s =
          wd0*(wh0*(ww0*xp[id0*2500+ih0*50+iw0] + ww1*xp[id0*2500+ih0*50+iw1])
             + wh1*(ww0*xp[id0*2500+ih1*50+iw0] + ww1*xp[id0*2500+ih1*50+iw1]))
        + wd1*(wh0*(ww0*xp[id1*2500+ih0*50+iw0] + ww1*xp[id1*2500+ih0*50+iw1])
             + wh1*(ww0*xp[id1*2500+ih1*50+iw0] + ww1*xp[id1*2500+ih1*50+iw1]));
        s *= mn;
        acc = fmaf(swdc[n], s, acc);
        if (p_out) {
            p_out[(size_t)idx*81 + n]      = pdc;
            p_out[(size_t)idx*81 + 27 + n] = phc;
            p_out[(size_t)idx*81 + 54 + n] = pwc;
        }
    }
    y_out[idx] = acc;
    if (xxx_out) xxx_out[idx] = acc;

    // block-reduce sum / sumsq for the scalar batchnorm
    red[tid] = acc; __syncthreads();
    for (int s = 128; s > 0; s >>= 1) { if (tid < s) red[tid] += red[tid+s]; __syncthreads(); }
    if (tid == 0) atomicAdd(&stat2[0], red[0]);
    __syncthreads();
    red[tid] = acc * acc; __syncthreads();
    for (int s = 128; s > 0; s >>= 1) { if (tid < s) red[tid] += red[tid+s]; __syncthreads(); }
    if (tid == 0) atomicAdd(&stat2[1], red[0]);
}

// whole-tensor BN + ReLU (C == 1)
__global__ void bn_relu_scalar_kernel(const float* __restrict__ y,
                                      const float* __restrict__ stat2,
                                      const float* __restrict__ g,
                                      const float* __restrict__ be,
                                      float* __restrict__ out) {
    int i = blockIdx.x * blockDim.x + threadIdx.x;
    if (i >= TOT) return;
    float inv = 1.f / (float)TOT;
    float mu  = stat2[0] * inv;
    float var = stat2[1] * inv - mu * mu;
    float sc  = g[0] * rsqrtf(var + EPSV);
    out[i] = fmaxf((y[i] - mu) * sc + be[0], 0.f);
}

// ---------------- conv c1: 1 -> 32 channels, 3^3, pad 1 ----------------
__global__ __launch_bounds__(256) void conv_c1_kernel(const float* __restrict__ pad,
                                                      const float* __restrict__ w,
                                                      float* __restrict__ h1) {
    __shared__ float sw[32*27];
    int tid = threadIdx.x;
    for (int i = tid; i < 32*27; i += 256) sw[i] = w[i];
    __syncthreads();
    int idx = blockIdx.x * 256 + tid;
    int xq = idx % DIMS, yq = (idx / DIMS) % DIMS,
        zq = (idx / (DIMS*DIMS)) % DIMS, bq = idx / VOL;
    const float* xp = pad + (size_t)bq * PVOL;
    float v[27];
    #pragma unroll
    for (int dz = 0; dz < 3; ++dz)
        #pragma unroll
        for (int dy = 0; dy < 3; ++dy)
            #pragma unroll
            for (int dx = 0; dx < 3; ++dx)
                v[dz*9 + dy*3 + dx] = xp[(zq+dz)*(PDIM*PDIM) + (yq+dy)*PDIM + (xq+dx)];
    int sp = zq*(DIMS*DIMS) + yq*DIMS + xq;
    for (int oc = 0; oc < 32; ++oc) {
        float a = 0.f;
        #pragma unroll
        for (int t = 0; t < 27; ++t) a = fmaf(sw[oc*27 + t], v[t], a);
        h1[((size_t)(bq*32 + oc))*VOL + sp] = a;
    }
}

// ---------------- per-channel BN statistics ----------------
__global__ void chanstats_kernel(const float* __restrict__ src,
                                 float* __restrict__ stat, int C) {
    __shared__ float red[256];
    int c = blockIdx.y, tid = threadIdx.x;
    float s = 0.f, s2 = 0.f;
    for (int t = blockIdx.x*256 + tid; t < TOT; t += gridDim.x*256) {
        int b = t / VOL, vv = t % VOL;
        float x = src[((size_t)(b*C + c))*VOL + vv];
        s += x; s2 = fmaf(x, x, s2);
    }
    red[tid] = s; __syncthreads();
    for (int k = 128; k > 0; k >>= 1) { if (tid < k) red[tid] += red[tid+k]; __syncthreads(); }
    if (tid == 0) atomicAdd(&stat[2*c], red[0]);
    __syncthreads();
    red[tid] = s2; __syncthreads();
    for (int k = 128; k > 0; k >>= 1) { if (tid < k) red[tid] += red[tid+k]; __syncthreads(); }
    if (tid == 0) atomicAdd(&stat[2*c + 1], red[0]);
}

// stats -> fused scale/bias:  out = h*scale[c] + bias[c]
__global__ void scalebias_kernel(const float* __restrict__ stat,
                                 const float* __restrict__ g,
                                 const float* __restrict__ be,
                                 float* __restrict__ sb, int C) {
    int c = blockIdx.x * blockDim.x + threadIdx.x;
    if (c >= C) return;
    float inv = 1.f / (float)TOT;
    float mu  = stat[2*c] * inv;
    float var = stat[2*c + 1] * inv - mu * mu;
    float sc  = g[c] * rsqrtf(var + EPSV);
    sb[c]     = sc;
    sb[C + c] = be[c] - mu * sc;
}

// BN+ReLU h1 -> padded (50^3) fp16 activation buffer for the WMMA GEMM
__global__ void prep_c2_kernel(const float* __restrict__ h1,
                               const float* __restrict__ sb,
                               _Float16* __restrict__ out) {
    int i = blockIdx.x * 256 + threadIdx.x;   // exactly NB*32*PVOL threads
    int x = i % PDIM, y = (i / PDIM) % PDIM, z = (i / (PDIM*PDIM)) % PDIM, bc = i / PVOL;
    int c = bc & 31;
    float v = 0.f;
    if (x > 0 && x < PDIM-1 && y > 0 && y < PDIM-1 && z > 0 && z < PDIM-1) {
        float h = h1[(size_t)bc * VOL + (z-1)*(DIMS*DIMS) + (y-1)*DIMS + (x-1)];
        v = fmaxf(fmaf(h, sb[c], sb[32 + c]), 0.f);
    }
    out[i] = (_Float16)v;
}

// convert w_c2 fp32 [64][32*27] -> fp16 [M=64][K=864]
__global__ void wprep_kernel(const float* __restrict__ w, _Float16* __restrict__ wh) {
    int i = blockIdx.x * 256 + threadIdx.x;
    if (i < 64 * KTOT) wh[i] = (_Float16)w[i];
}

// ---------------- WMMA GEMM: conv c2 (32 -> 64 channels) ----------------
// D[64 x 16pts] = W[64 x 864] * im2col[864 x 16pts], K chunks of 32.
// Block = 128 threads = 4 waves; wave m owns output channels 16m..16m+15.
__device__ __forceinline__ _Float16 ldB(const _Float16* __restrict__ base,
                                        int k, int z, int y, int x) {
    int c = k / 27, tap = k - c * 27;
    int dz = tap / 9, dy = (tap / 3) % 3, dx = tap % 3;
    return base[(size_t)c * PVOL + (z + dz)*(PDIM*PDIM) + (y + dy)*PDIM + (x + dx)];
}

__global__ __launch_bounds__(128) void wmma_c2_kernel(const _Float16* __restrict__ h1rp,
                                                      const _Float16* __restrict__ wA,
                                                      float* __restrict__ h2) {
    int tile = blockIdx.x;                 // NB*48*48*3 tiles
    int xt = tile % 3;
    int y  = (tile / 3) % DIMS;
    int z  = (tile / 144) % DIMS;
    int b  = tile / (144 * DIMS);
    int wave = threadIdx.x >> 5;
    int lane = threadIdx.x & 31;
    int nIdx = lane & 15;                  // column (output point) index
    int half = lane >> 4;
    int xcoord = xt * 16 + nIdx;
    const _Float16* bbase = h1rp + (size_t)b * 32 * PVOL;
    const _Float16* arow  = wA + (size_t)(wave * 16 + nIdx) * KTOT;  // A: lane%16 = M row

    v8f acc = {};
    for (int j = 0; j < 27; ++j) {
        int kbase = j * 32;
        __builtin_prefetch(arow + kbase + 64, 0, 3);   // global_prefetch_b8 on weight stream
        v16h a, bf;
        #pragma unroll
        for (int i = 0; i < 8; ++i) {
            // A fragment (16-bit A 16x32 ISA layout): VGPR i<4 -> K=2i+8*half,
            // i>=4 -> K=2i+8+8*half; pairs (K, K+1)
            int kloA = (i < 4) ? (2*i + 8*half) : (2*i + 8 + 8*half);
            int kA0 = kbase + kloA;
            a[2*i]     = arow[kA0];
            a[2*i + 1] = arow[kA0 + 1];
            // B fragment: same K mapping as A
            int kB0 = kA0;
            bf[2*i]     = ldB(bbase, kB0,     z, y, xcoord);
            bf[2*i + 1] = ldB(bbase, kB0 + 1, z, y, xcoord);
        }
        acc = __builtin_amdgcn_wmma_f32_16x16x32_f16(
                  false, a, false, bf, (short)0, acc, false, false);
    }
    // D layout: VGPR jj holds row M = jj + 8*half, col N = lane%16
    #pragma unroll
    for (int jj = 0; jj < 8; ++jj) {
        int oc = wave * 16 + jj + 8 * half;
        h2[((size_t)(b * 64 + oc))*VOL + z*(DIMS*DIMS) + y*DIMS + xcoord] = acc[jj];
    }
}

// ---------------- BN+ReLU + 3^3 maxpool + global mean ----------------
__global__ __launch_bounds__(256) void pool_mean_kernel(const float* __restrict__ h2,
                                                        const float* __restrict__ sb,
                                                        float* __restrict__ hbar) {
    __shared__ float red[256];
    int bc = blockIdx.x;                   // 0..255 = b*64 + c
    int c = bc & 63, tid = threadIdx.x;
    float sc = sb[c], bi = sb[64 + c];
    const float* src = h2 + (size_t)bc * VOL;
    float sum = 0.f;
    for (int cell = tid; cell < 16*16*16; cell += 256) {
        int px = cell & 15, py = (cell >> 4) & 15, pz = cell >> 8;
        float mx = -INFINITY;
        #pragma unroll
        for (int dz = 0; dz < 3; ++dz)
            #pragma unroll
            for (int dy = 0; dy < 3; ++dy)
                #pragma unroll
                for (int dx = 0; dx < 3; ++dx) {
                    float h = src[(pz*3 + dz)*(DIMS*DIMS) + (py*3 + dy)*DIMS + (px*3 + dx)];
                    mx = fmaxf(mx, fmaxf(fmaf(h, sc, bi), 0.f));
                }
        sum += mx;
    }
    red[tid] = sum; __syncthreads();
    for (int k = 128; k > 0; k >>= 1) { if (tid < k) red[tid] += red[tid+k]; __syncthreads(); }
    if (tid == 0) hbar[bc] = red[0] * (1.f / 4096.f);
}

__global__ void fc_kernel(const float* __restrict__ hbar, const float* __restrict__ wfc,
                          const float* __restrict__ bfc, float* __restrict__ out) {
    int t = threadIdx.x;
    if (t >= 40) return;
    int b = t / 10, o = t % 10;
    float a = bfc[o];
    #pragma unroll
    for (int i = 0; i < 64; ++i) a = fmaf(hbar[b*64 + i], wfc[o*64 + i], a);
    out[b*10 + o] = a;
}

// ---------------- host launcher ----------------
extern "C" void kernel_launch(void* const* d_in, const int* in_sizes, int n_in,
                              void* d_out, int out_size, void* d_ws, size_t ws_size,
                              hipStream_t stream) {
    const float* x    = (const float*)d_in[0];
    const float* w_p  = (const float*)d_in[1];
    const float* b_p  = (const float*)d_in[2];
    const float* w_m  = (const float*)d_in[3];
    const float* b_m  = (const float*)d_in[4];
    const float* w_dc = (const float*)d_in[5];
    const float* g_d  = (const float*)d_in[6];
    const float* be_d = (const float*)d_in[7];
    const float* w_c1 = (const float*)d_in[8];
    const float* g1   = (const float*)d_in[9];
    const float* be1  = (const float*)d_in[10];
    const float* w_c2 = (const float*)d_in[11];
    const float* g2   = (const float*)d_in[12];
    const float* be2  = (const float*)d_in[13];
    const float* w_fc = (const float*)d_in[14];
    const float* b_fc = (const float*)d_in[15];

    float* out   = (float*)d_out;
    float* xxx   = out + 40;
    float* p_out = out + 40 + TOT;

    char* base = (char*)d_ws;
    size_t off = 0;
    auto alloc = [&](size_t bytes) -> void* {
        void* p = base + off;
        off = (off + bytes + 255) & ~(size_t)255;
        return p;
    };
    float*    padb  = (float*)   alloc(sizeof(float)   * (size_t)NB * PVOL);
    float*    ybuf  = (float*)   alloc(sizeof(float)   * (size_t)TOT);
    float*    inpb  = (float*)   alloc(sizeof(float)   * (size_t)TOT);
    float*    h1    = (float*)   alloc(sizeof(float)   * (size_t)NB * 32 * VOL);
    _Float16* h1rp  = (_Float16*)alloc(sizeof(_Float16)* (size_t)NB * 32 * PVOL);
    float*    h2    = (float*)   alloc(sizeof(float)   * (size_t)NB * 64 * VOL);
    _Float16* wA    = (_Float16*)alloc(sizeof(_Float16)* 64 * KTOT);
    float*    stats = (float*)   alloc(sizeof(float)   * 256);
    float*    sb1   = (float*)   alloc(sizeof(float)   * 64);
    float*    sb2   = (float*)   alloc(sizeof(float)   * 128);
    float*    hbar  = (float*)   alloc(sizeof(float)   * 256);
    (void)ws_size; (void)in_sizes; (void)n_in; (void)out_size;

    float* yst0 = stats;        // [0..1]
    float* yst1 = stats + 2;    // [2..3]
    float* st1  = stats + 16;   // 64 floats
    float* st2  = stats + 96;   // 128 floats

    const int padBlocks = (NB * PVOL + 255) / 256;   // 1954
    const int ptBlocks  = TOT / 256;                 // 1728 (exact)

    zero_kernel<<<1, 256, 0, stream>>>(stats, 256);
    wprep_kernel<<<(64*KTOT)/256, 256, 0, stream>>>(w_c2, wA);

    // deform iteration 0 (input = x)
    pad_kernel<<<padBlocks, 256, 0, stream>>>(x, padb);
    deform_kernel<<<ptBlocks, 256, 0, stream>>>(padb, w_p, b_p, w_m, b_m, w_dc,
                                                ybuf, yst0, nullptr, nullptr);
    bn_relu_scalar_kernel<<<ptBlocks, 256, 0, stream>>>(ybuf, yst0, g_d, be_d, inpb);

    // deform iteration 1 (input = relu(bn(y0))); emits xxx and p
    pad_kernel<<<padBlocks, 256, 0, stream>>>(inpb, padb);
    deform_kernel<<<ptBlocks, 256, 0, stream>>>(padb, w_p, b_p, w_m, b_m, w_dc,
                                                ybuf, yst1, xxx, p_out);
    bn_relu_scalar_kernel<<<ptBlocks, 256, 0, stream>>>(ybuf, yst1, g_d, be_d, inpb);

    // conv c1 (1->32) + per-channel BN stats
    pad_kernel<<<padBlocks, 256, 0, stream>>>(inpb, padb);
    conv_c1_kernel<<<ptBlocks, 256, 0, stream>>>(padb, w_c1, h1);
    chanstats_kernel<<<dim3(54, 32), 256, 0, stream>>>(h1, st1, 32);
    scalebias_kernel<<<1, 64, 0, stream>>>(st1, g1, be1, sb1, 32);

    // BN+ReLU -> padded fp16 im2col source, then WMMA GEMM (32 -> 64)
    prep_c2_kernel<<<(NB*32*PVOL)/256, 256, 0, stream>>>(h1, sb1, h1rp);
    wmma_c2_kernel<<<NB * DIMS * DIMS * 3, 128, 0, stream>>>(h1rp, wA, h2);

    // per-channel BN stats for h2, then pool+mean+fc
    chanstats_kernel<<<dim3(54, 64), 256, 0, stream>>>(h2, st2, 64);
    scalebias_kernel<<<1, 64, 0, stream>>>(st2, g2, be2, sb2, 64);
    pool_mean_kernel<<<NB * 64, 256, 0, stream>>>(h2, sb2, hbar);
    fc_kernel<<<1, 64, 0, stream>>>(hbar, w_fc, b_fc, out);
}
